// MultiHeadAttention__5987184410841
// MI455X (gfx1250) — matrix operand order repaired and checked by reading, not compiled, not run on hardware
//
#include <hip/hip_runtime.h>

// MI455X / gfx1250 fused multi-head attention with head-axis softmax quirk.
// bf16 WMMA (v_wmma_f32_16x16x32_bf16) + TDM (tensor_load_to_lds) double-buffered
// LDS tiling for the projection/output GEMMs.

typedef __bf16 bf16_t;
typedef __attribute__((ext_vector_type(16))) __bf16 v16bf;
typedef __attribute__((ext_vector_type(8)))  float  v8f;
typedef unsigned int u32;
typedef unsigned int v4u __attribute__((ext_vector_type(4)));
typedef int          v8i __attribute__((ext_vector_type(8)));
typedef int          v4i __attribute__((ext_vector_type(4)));

#define BB 2
#define TT 2048
#define DD 1024
#define HH 16
// HD = 64

#ifndef __has_builtin
#define __has_builtin(x) 0
#endif
#if __has_builtin(__builtin_amdgcn_tensor_load_to_lds)
#define HAVE_TDM 1
#if __has_include(<hip/amd_detail/amd_gfx1250_TDM.h>)
#define TDM_6ARG 1
#else
#define TDM_6ARG 0
#endif
#else
#define HAVE_TDM 0
#endif

union FragAB { v16bf v; uint4 q[2]; };   // 16 bf16 = 32B = 8 VGPRs
union FragC  { v8f  v; float f[8]; };    // 16x16 f32 C/D tile slice per lane

__device__ __forceinline__ u32 lds_off(const void* p) {
  return (u32)(uintptr_t)p;   // generic LDS addr low 32 bits == LDS byte offset
}

#if HAVE_TDM
// Issue one TDM descriptor: load a rows x 32 bf16 tile (row stride K=1024 elems)
// from global into LDS, packed row-major [row][32]. D# per CDNA5 ISA ch.8.
__device__ __forceinline__ void tdm_load_tile(const bf16_t* g, u32 lds_addr, u32 rows) {
  unsigned long long ga = (unsigned long long)(uintptr_t)g;
  v4u g0;
  g0.x = 1u;                                              // count=1, no gather
  g0.y = lds_addr;                                        // lds_addr [63:32]
  g0.z = (u32)ga;                                         // global_addr[31:0]
  g0.w = ((u32)(ga >> 32) & 0x01FFFFFFu) | (2u << 30);    // addr[56:32] | type=2
  v8i g1;
  g1[0] = 0x00010000;               // wg_mask=0, data_size=1 (2 bytes)
  g1[1] = 0;                        // atomic_barrier=0, tensor_dim0 lo16 (dim0=1<<20)
  g1[2] = 0x10;                     // tensor_dim0 hi16 = 0x10, tensor_dim1 lo16 = 0
  g1[3] = (32 << 16) | 0x10;        // tensor_dim1 hi16 = 0x10, tile_dim0 = 32
  g1[4] = (int)rows;                // tile_dim1 = rows, tile_dim2 = 0
  g1[5] = 1024;                     // tensor_dim0_stride = K elements
  g1[6] = 0;                        // stride hi, tensor_dim1_stride lo
  g1[7] = 0;                        // tensor_dim1_stride hi
  v4i zz = {0, 0, 0, 0};
#if TDM_6ARG
  v8i z8 = {0, 0, 0, 0, 0, 0, 0, 0};
  __builtin_amdgcn_tensor_load_to_lds(g0, g1, zz, zz, z8, 0);
#else
  __builtin_amdgcn_tensor_load_to_lds(g0, g1, zz, zz, 0);
#endif
}
#endif

// ---------------------------------------------------------------- cast x -> bf16
__global__ __launch_bounds__(256) void mha_cast_f32_bf16(const float* __restrict__ in,
                                                         bf16_t* __restrict__ out, int n) {
  int i = (blockIdx.x * blockDim.x + threadIdx.x) * 4;
  if (i < n) {
    float4 v = *(const float4*)(in + i);
    out[i + 0] = (bf16_t)v.x;
    out[i + 1] = (bf16_t)v.y;
    out[i + 2] = (bf16_t)v.z;
    out[i + 3] = (bf16_t)v.w;
  }
}

// --------------------------------------------- transpose+cast weight: Wt[n][k] = W[k][n]
__global__ __launch_bounds__(256) void mha_transpose_cast(const float* __restrict__ W,
                                                          bf16_t* __restrict__ Wt) {
  int idx = blockIdx.x * blockDim.x + threadIdx.x;   // over DD*DD
  int k = idx >> 10;
  int n = idx & 1023;
  Wt[(n << 10) | k] = (bf16_t)W[idx];
}

// ---------------------------------------------------------------- WMMA GEMM (TDM-tiled)
// C[M=4096][N=1024] = A[M][K=1024](bf16,row) @ Wt[N][K](bf16, pre-transposed) + bias
// Block tile 64(M) x 128(N), 8 waves of 32x32; K-steps of 32 double-buffered via TDM.
// mode 0: bf16 out [m][n]; mode 1: bf16 out V-transposed; mode 2: f32 out [m][n]
__global__ __launch_bounds__(256) void mha_gemm_bf16(const bf16_t* __restrict__ A,
                                                     const bf16_t* __restrict__ Wt,
                                                     const float* __restrict__ bias,
                                                     void* __restrict__ out, int mode) {
  const int K = DD, N = DD;
  __shared__ bf16_t sA[2][64 * 32];    // 4 KB x2
  __shared__ bf16_t sB[2][128 * 32];   // 8 KB x2

  const int lane = threadIdx.x & 31;
  const int wave = threadIdx.x >> 5;
  const int hi  = lane >> 4;           // 0: lanes 0-15, 1: lanes 16-31
  const int l16 = lane & 15;
  const int m0blk = blockIdx.x * 64;
  const int n0blk = blockIdx.y * 128;
  const int wm = (wave >> 2) * 32;     // wave M offset within block tile
  const int wn = (wave & 3) * 32;      // wave N offset within block tile

  FragC acc[2][2];
#pragma unroll
  for (int i = 0; i < 2; i++)
#pragma unroll
    for (int j = 0; j < 2; j++)
#pragma unroll
      for (int e = 0; e < 8; e++) acc[i][j].f[e] = 0.f;

#if HAVE_TDM
  if (wave == 0) {   // prologue: fill buffer 0
    tdm_load_tile(A  + (size_t)m0blk * K, lds_off(&sA[0][0]), 64);
    tdm_load_tile(Wt + (size_t)n0blk * K, lds_off(&sB[0][0]), 128);
  }
#endif

  for (int ks = 0; ks < K; ks += 32) {
    const int buf = (ks >> 5) & 1;
#if HAVE_TDM
    if (wave == 0) {
      if (ks + 32 < K) {   // prefetch next tiles into other buffer
        tdm_load_tile(A  + (size_t)m0blk * K + (ks + 32), lds_off(&sA[buf ^ 1][0]), 64);
        tdm_load_tile(Wt + (size_t)n0blk * K + (ks + 32), lds_off(&sB[buf ^ 1][0]), 128);
        __builtin_amdgcn_s_wait_tensorcnt(2);   // current buffer's pair complete
      } else {
        __builtin_amdgcn_s_wait_tensorcnt(0);
      }
    }
    __syncthreads();
#else
    __syncthreads();   // protect previous reads before refill
    {
      const int tid = threadIdx.x;
      {  // A: 64x32 = 256 uint4
        int r = tid >> 2, c = (tid & 3) * 8;
        *(uint4*)&sA[buf][r * 32 + c] = *(const uint4*)(A + (size_t)(m0blk + r) * K + ks + c);
      }
#pragma unroll
      for (int t = 0; t < 2; t++) {   // B: 128x32 = 512 uint4
        int idx = tid + t * 256;
        int r = idx >> 2, c = (idx & 3) * 8;
        *(uint4*)&sB[buf][r * 32 + c] = *(const uint4*)(Wt + (size_t)(n0blk + r) * K + ks + c);
      }
    }
    __syncthreads();
#endif

    FragAB a[2], b[2];
    // A-frag (16x32 MxK): row = wm+tM*16+l16; elems 0-7 -> K=8*hi.., 8-15 -> 16+8*hi
#pragma unroll
    for (int tM = 0; tM < 2; tM++) {
      const bf16_t* ap = &sA[buf][(wm + tM * 16 + l16) * 32];
      a[tM].q[0] = *(const uint4*)(ap + 8 * hi);
      a[tM].q[1] = *(const uint4*)(ap + 16 + 8 * hi);
    }
    // B-frag (32x16 KxN): col = wn+tN*16+l16; elems 0-7 -> K=8*hi.., 8-15 -> 16+8*hi
#pragma unroll
    for (int tN = 0; tN < 2; tN++) {
      const bf16_t* bp = &sB[buf][(wn + tN * 16 + l16) * 32 + 8 * hi];
      b[tN].q[0] = *(const uint4*)(bp);
      b[tN].q[1] = *(const uint4*)(bp + 16);
    }
#pragma unroll
    for (int tM = 0; tM < 2; tM++)
#pragma unroll
      for (int tN = 0; tN < 2; tN++)
        acc[tM][tN].v = __builtin_amdgcn_wmma_f32_16x16x32_bf16(
            false, a[tM].v, false, b[tN].v, (short)0, acc[tM][tN].v, false, false);
#if HAVE_TDM
    __syncthreads();   // readers done before next TDM overwrites this buffer
#endif
  }

#pragma unroll
  for (int tM = 0; tM < 2; tM++)
#pragma unroll
    for (int tN = 0; tN < 2; tN++)
#pragma unroll
      for (int r = 0; r < 8; r++) {
        int m = m0blk + wm + tM * 16 + r + hi * 8;   // C/D: M = r + 8*(lane>=16)
        int n = n0blk + wn + tN * 16 + l16;          //      N = lane%16
        float v = acc[tM][tN].f[r] + bias[n];
        if (mode == 0) {
          ((bf16_t*)out)[(size_t)m * N + n] = (bf16_t)v;
        } else if (mode == 1) {
          int bb = m >> 11, t = m & (TT - 1);
          ((bf16_t*)out)[((size_t)bb * N + n) * TT + t] = (bf16_t)v;
        } else {
          ((float*)out)[(size_t)m * N + n] = v;
        }
      }
}

// ---------------------------------------------------------------- fused attention
// Head-axis softmax: weight[b,h,i,j] = softmax over h of masked score.
// Masked (j>i) entries are -1e10 in ALL heads -> weight = 1/16 (faithful quirk),
// so PV runs over the FULL j range. Workgroup = 16 waves = 16 heads, one 16-row i-tile.
__global__ __launch_bounds__(512) void mha_attn_headsoftmax(
    const bf16_t* __restrict__ Q,   // [b][t][h*64+hd]
    const bf16_t* __restrict__ Km,  // [b][t][h*64+hd]
    const bf16_t* __restrict__ Vt,  // [b][h*64+hd][t]  (pre-transposed)
    bf16_t* __restrict__ Ao) {      // [b][t][h*64+hd]
  __shared__ float  sS[16 * 32 * 16];   // [i_local][j_local(32)][h]  32 KB
  __shared__ bf16_t sW[16 * 16 * 32];   // [h][i_local][j_local(32)]  16 KB

  const int lane = threadIdx.x & 31;
  const int h    = threadIdx.x >> 5;    // wave == head
  const int hi   = lane >> 4;
  const int l16  = lane & 15;
  const int b    = blockIdx.y;
  const int i0   = blockIdx.x * 16;

  // Q A-frags for this head, K-dim = hd 0..63 -> two K=32 frags, resident all kernel
  FragAB qf[2];
  {
    const bf16_t* qb = Q + ((size_t)(b * TT) + i0 + l16) * DD + h * 64;
#pragma unroll
    for (int f = 0; f < 2; f++) {
      qf[f].q[0] = *(const uint4*)(qb + f * 32 + hi * 8);
      qf[f].q[1] = *(const uint4*)(qb + f * 32 + 16 + hi * 8);
    }
  }

  FragC o[4];   // O[i 16][hd 64] accumulators
#pragma unroll
  for (int tN = 0; tN < 4; tN++)
#pragma unroll
    for (int e = 0; e < 8; e++) o[tN].f[e] = 0.f;

  const float scale = 0.03125f;  // 1/sqrt(1024)

  for (int j0 = 0; j0 < TT; j0 += 32) {
    // ---- scores: two 16-col subtiles, each K=64 -> 2 wmma
    FragC s[2];
#pragma unroll
    for (int sub = 0; sub < 2; sub++) {
#pragma unroll
      for (int e = 0; e < 8; e++) s[sub].f[e] = 0.f;
      const bf16_t* kb = Km + ((size_t)(b * TT) + j0 + sub * 16 + l16) * DD + h * 64;
#pragma unroll
      for (int f = 0; f < 2; f++) {
        FragAB bfr;   // B-frag [d][j]: elems 0-7 -> d = f*32 + 8*hi.., 8-15 -> f*32 + 16 + 8*hi
        const bf16_t* bp = kb + f * 32 + hi * 8;
        bfr.q[0] = *(const uint4*)(bp);
        bfr.q[1] = *(const uint4*)(bp + 16);
        s[sub].v = __builtin_amdgcn_wmma_f32_16x16x32_bf16(
            false, qf[f].v, false, bfr.v, (short)0, s[sub].v, false, false);
      }
    }
    // ---- scale + causal(-1e10) + scatter to LDS [i][j][h]
#pragma unroll
    for (int sub = 0; sub < 2; sub++)
#pragma unroll
      for (int r = 0; r < 8; r++) {
        int il = r + hi * 8;
        int jl = sub * 16 + l16;
        float v = s[sub].f[r] * scale;
        if ((j0 + jl) > (i0 + il)) v = -1e10f;
        sS[(il * 32 + jl) * 16 + h] = v;
      }
    __syncthreads();
    // ---- head softmax: 512 threads == 16x32 (i,j) pairs, 16-wide reduce each
    {
      int il = threadIdx.x >> 5, jl = threadIdx.x & 31;
      const float* p = &sS[(il * 32 + jl) * 16];
      float m = p[0];
#pragma unroll
      for (int t = 1; t < 16; t++) m = fmaxf(m, p[t]);
      float ex[16], sum = 0.f;
#pragma unroll
      for (int t = 0; t < 16; t++) { ex[t] = __expf(p[t] - m); sum += ex[t]; }
      float rs = 1.f / sum;   // all-masked column -> exp(0)*16 -> 1/16 each (faithful)
#pragma unroll
      for (int t = 0; t < 16; t++)
        sW[(t * 16 + il) * 32 + jl] = (bf16_t)(ex[t] * rs);
    }
    __syncthreads();
    // ---- PV: A = weights (16 x K=32 over j), B = Vt (j x hd), 4 wmma
    FragAB wf;
    {
      const bf16_t* wp = &sW[(h * 16 + l16) * 32];
      wf.q[0] = *(const uint4*)(wp + hi * 8);
      wf.q[1] = *(const uint4*)(wp + 16 + hi * 8);
    }
#pragma unroll
    for (int tN = 0; tN < 4; tN++) {
      FragAB vf;  // elems 0-7 -> j = j0 + 8*hi.., 8-15 -> j0 + 16 + 8*hi
      const bf16_t* vp = Vt + ((size_t)(b * DD) + h * 64 + tN * 16 + l16) * TT + j0 + hi * 8;
      vf.q[0] = *(const uint4*)(vp);
      vf.q[1] = *(const uint4*)(vp + 16);
      o[tN].v = __builtin_amdgcn_wmma_f32_16x16x32_bf16(
          false, wf.v, false, vf.v, (short)0, o[tN].v, false, false);
    }
    // no barrier needed: sS rewrites next iter are fenced by the barrier after sW writes
  }

  // ---- store attention output (bf16) in [b][t][h*64+hd]
#pragma unroll
  for (int tN = 0; tN < 4; tN++)
#pragma unroll
    for (int r = 0; r < 8; r++) {
      int ig = i0 + r + hi * 8;
      int d  = h * 64 + tN * 16 + l16;
      Ao[((size_t)(b * TT) + ig) * DD + d] = (bf16_t)o[tN].f[r];
    }
}

// ---------------------------------------------------------------- launcher
extern "C" void kernel_launch(void* const* d_in, const int* in_sizes, int n_in,
                              void* d_out, int out_size, void* d_ws, size_t ws_size,
                              hipStream_t stream) {
  const float* x  = (const float*)d_in[0];
  const float* Wq = (const float*)d_in[1];
  const float* bq = (const float*)d_in[2];
  const float* Wk = (const float*)d_in[3];
  const float* bk = (const float*)d_in[4];
  const float* Wv = (const float*)d_in[5];
  const float* bv = (const float*)d_in[6];
  const float* Wo = (const float*)d_in[7];
  const float* bo = (const float*)d_in[8];

  char* ws = (char*)d_ws;
  bf16_t* xbf = (bf16_t*)(ws + (size_t)0);          // 8 MB
  bf16_t* Wtq = (bf16_t*)(ws + ((size_t)8  << 20)); // 2 MB each
  bf16_t* Wtk = (bf16_t*)(ws + ((size_t)10 << 20));
  bf16_t* Wtv = (bf16_t*)(ws + ((size_t)12 << 20));
  bf16_t* Wto = (bf16_t*)(ws + ((size_t)14 << 20));
  bf16_t* Qb  = (bf16_t*)(ws + ((size_t)16 << 20)); // 8 MB
  bf16_t* Kb  = (bf16_t*)(ws + ((size_t)24 << 20)); // 8 MB
  bf16_t* Vt  = (bf16_t*)(ws + ((size_t)32 << 20)); // 8 MB (transposed)
  bf16_t* Ab  = (bf16_t*)(ws + ((size_t)40 << 20)); // 8 MB

  const int nx = BB * TT * DD;                        // 4,194,304
  mha_cast_f32_bf16<<<nx / 4 / 256, 256, 0, stream>>>(x, xbf, nx);

  const int nw = DD * DD;                             // 1,048,576
  mha_transpose_cast<<<nw / 256, 256, 0, stream>>>(Wq, Wtq);
  mha_transpose_cast<<<nw / 256, 256, 0, stream>>>(Wk, Wtk);
  mha_transpose_cast<<<nw / 256, 256, 0, stream>>>(Wv, Wtv);
  mha_transpose_cast<<<nw / 256, 256, 0, stream>>>(Wo, Wto);

  dim3 gg(BB * TT / 64, DD / 128);                    // 64 x 8 blocks
  mha_gemm_bf16<<<gg, 256, 0, stream>>>(xbf, Wtq, bq, (void*)Qb, 0);
  mha_gemm_bf16<<<gg, 256, 0, stream>>>(xbf, Wtk, bk, (void*)Kb, 0);
  mha_gemm_bf16<<<gg, 256, 0, stream>>>(xbf, Wtv, bv, (void*)Vt, 1);

  dim3 ga(TT / 16, BB);                               // 128 x 2 blocks, 512 thr
  mha_attn_headsoftmax<<<ga, 512, 0, stream>>>(Qb, Kb, Vt, Ab);

  mha_gemm_bf16<<<gg, 256, 0, stream>>>(Ab, Wto, bo, d_out, 2);
}
